// PanelAgentGNN_22694607192566
// MI455X (gfx1250) — matrix operand order repaired and checked by reading, not compiled, not run on hardware
//
#include <hip/hip_runtime.h>
#include <hip/hip_bf16.h>

// ---------------- sizes (fixed by the reference) ----------------
#define NN   6144
#define NQQ  3072
#define DD   128
#define HH   4
#define CC   32
#define FIN  10
#define LL   4
#define EE   393216
#define ETOT (EE + NN)          // edges + self loops = 399360

// monotonic encoding of -inf for float-as-uint atomicMax init
#define FENC_NEG_INF 0x007FFFFFu

// ---------------- WMMA helper types ----------------
typedef __bf16 bf16_t;
typedef bf16_t v16bf __attribute__((ext_vector_type(16)));
typedef float  v8f   __attribute__((ext_vector_type(8)));
typedef unsigned short u16v16 __attribute__((ext_vector_type(16)));

__device__ __forceinline__ unsigned short f2bf(float f) {
    unsigned u = __float_as_uint(f);
    u += 0x7FFFu + ((u >> 16) & 1u);   // round-to-nearest-even
    return (unsigned short)(u >> 16);
}

__device__ __forceinline__ v8f wmma_bf16(const u16v16& a, const u16v16& b, v8f c) {
    v16bf av = __builtin_bit_cast(v16bf, a);
    v16bf bv = __builtin_bit_cast(v16bf, b);
    return __builtin_amdgcn_wmma_f32_16x16x32_bf16(false, av, false, bv,
                                                   (short)0, c, false, false);
}

// monotonic float<->uint mapping for atomicMax on floats
__device__ __forceinline__ unsigned fenc(float f) {
    unsigned u = __float_as_uint(f);
    return (u & 0x80000000u) ? ~u : (u | 0x80000000u);
}
__device__ __forceinline__ float fdec(unsigned e) {
    return (e & 0x80000000u) ? __uint_as_float(e & 0x7FFFFFFFu)
                             : __uint_as_float(~e);
}

// ---------------- block-of-128 LayerNorm helpers ----------------
__device__ __forceinline__ float block_sum_128(float v, float* red) {
    int c = threadIdx.x;
    red[c] = v; __syncthreads();
    for (int off = 64; off > 0; off >>= 1) {
        if (c < off) red[c] += red[c + off];
        __syncthreads();
    }
    float r = red[0]; __syncthreads();
    return r;
}
__device__ __forceinline__ float ln128(float v, const float* g, const float* b, float* red) {
    int c = threadIdx.x;
    float mean = block_sum_128(v, red) * (1.f / 128.f);
    float d = v - mean;
    float var = block_sum_128(d * d, red) * (1.f / 128.f);
    return d * rsqrtf(var + 1e-5f) * g[c] + b[c];
}

// ================= embedding: x = relu(LN(nf @ W + b)) =================
__global__ __launch_bounds__(128) void k_embed(const float* __restrict__ nf,
                                               const float* __restrict__ W,
                                               const float* __restrict__ bias,
                                               const float* __restrict__ g,
                                               const float* __restrict__ bb,
                                               float* __restrict__ X) {
    __shared__ float red[128];
    int n = blockIdx.x, c = threadIdx.x;
    float acc = bias[c];
    #pragma unroll
    for (int f = 0; f < FIN; ++f) acc += nf[n * FIN + f] * W[f * DD + c];
    float y = ln128(acc, g, bb, red);
    X[n * DD + c] = fmaxf(y, 0.f);
}

// ================= generic Y = act(X[M x 128] @ W[128 x NCOLS] + bias) =================
// 4 waves / block, 64 rows / block. Weights + X tile staged in LDS as bf16.
// NCOLS is compile-time so the accumulator array fully unrolls into VGPRs (no scratch).
template <int NCOLS>
__global__ __launch_bounds__(128) void k_gemm(const float* __restrict__ X,
                                              const float* __restrict__ W,
                                              const float* __restrict__ bias,
                                              float* __restrict__ Y,
                                              int do_relu) {
    __shared__ unsigned short sW[128 * NCOLS];
    __shared__ unsigned short sX[64 * 128];
    const int tid  = threadIdx.x;
    const int row0 = blockIdx.x * 64;

    #pragma unroll 4
    for (int i = tid; i < 128 * NCOLS; i += 128) sW[i] = f2bf(W[i]);
    #pragma unroll 4
    for (int i = tid; i < 64 * 128;    i += 128)
        sX[i] = f2bf(X[(row0 + (i >> 7)) * DD + (i & 127)]);
    __syncthreads();

    const int wave = tid >> 5, lane = tid & 31;
    const int m = lane & 15, half = lane >> 4;
    constexpr int NT = NCOLS / 16;

    v8f acc[NT];
    v8f zero = {0.f,0.f,0.f,0.f,0.f,0.f,0.f,0.f};
    #pragma unroll
    for (int t = 0; t < NT; ++t) acc[t] = zero;

    #pragma unroll
    for (int kt = 0; kt < 4; ++kt) {
        const int k0 = kt * 32;
        // A fragment (16x32 bf16, CDNA5 layout)
        u16v16 a;
        const unsigned short* xr = &sX[(wave * 16 + m) * 128 + k0];
        #pragma unroll
        for (int j = 0; j < 8; ++j) {
            int kk = ((j < 4) ? 0 : 16) + half * 8 + 2 * (j & 3);
            a[2 * j]     = xr[kk];
            a[2 * j + 1] = xr[kk + 1];
        }
        #pragma unroll
        for (int ct = 0; ct < NT; ++ct) {
            // B fragment (32x16 bf16): lane supplies N column, slots = K rows (table order)
            u16v16 b;
            const unsigned short* wr = &sW[k0 * NCOLS + ct * 16 + m];
            #pragma unroll
            for (int n = 0; n < 16; ++n) b[n] = wr[(((n < 8) ? 0 : 16) + half * 8 + (n & 7)) * NCOLS];
            acc[ct] = wmma_bf16(a, b, acc[ct]);
        }
    }
    #pragma unroll
    for (int ct = 0; ct < NT; ++ct) {
        int col = ct * 16 + m;
        float bv = bias ? bias[col] : 0.f;
        #pragma unroll
        for (int r = 0; r < 8; ++r) {
            int row = row0 + wave * 16 + r + half * 8;
            float v = acc[ct][r] + bv;
            if (do_relu) v = fmaxf(v, 0.f);
            Y[row * NCOLS + col] = v;
        }
    }
}

// ================= GAT attention scores a_s, a_d =================
__global__ void k_scores(const float* __restrict__ Hf,
                         const float* __restrict__ asrc,
                         const float* __restrict__ adst,
                         float* __restrict__ AS, float* __restrict__ AD) {
    int i = blockIdx.x * blockDim.x + threadIdx.x;
    if (i >= NN * HH) return;
    int n = i >> 2, h = i & 3;
    const float* hr = Hf + n * DD + h * CC;
    float s = 0.f, d = 0.f;
    #pragma unroll
    for (int c = 0; c < CC; ++c) { s += hr[c] * asrc[h * CC + c]; d += hr[c] * adst[h * CC + c]; }
    AS[i] = s; AD[i] = d;
}

// ================= fills =================
__global__ void k_fill_f(float* p, float v, int n) {
    for (int i = blockIdx.x * blockDim.x + threadIdx.x; i < n; i += gridDim.x * blockDim.x) p[i] = v;
}
__global__ void k_fill_u(unsigned* p, unsigned v, int n) {
    for (int i = blockIdx.x * blockDim.x + threadIdx.x; i < n; i += gridDim.x * blockDim.x) p[i] = v;
}

// ================= edge phase (segment softmax) =================
__device__ __forceinline__ void edge_sd(const int* ei, int e, int& s, int& d) {
    if (e < EE) { s = ei[e]; d = ei[EE + e]; } else { s = e - EE; d = e - EE; }
}
__global__ void k_edge_max(const int* __restrict__ ei,
                           const float* __restrict__ AS, const float* __restrict__ AD,
                           unsigned* __restrict__ MX) {
    for (long i = blockIdx.x * (long)blockDim.x + threadIdx.x; i < (long)ETOT * HH;
         i += (long)gridDim.x * blockDim.x) {
        int e = (int)(i >> 2), h = (int)(i & 3), s, d;
        edge_sd(ei, e, s, d);
        float sc = AS[s * HH + h] + AD[d * HH + h];
        sc = sc > 0.f ? sc : 0.2f * sc;                 // leaky_relu 0.2
        atomicMax(&MX[d * HH + h], fenc(sc));
    }
}
__global__ void k_edge_expsum(const int* __restrict__ ei,
                              const float* __restrict__ AS, const float* __restrict__ AD,
                              const unsigned* __restrict__ MX,
                              float* __restrict__ EX, float* __restrict__ DEN) {
    for (long i = blockIdx.x * (long)blockDim.x + threadIdx.x; i < (long)ETOT * HH;
         i += (long)gridDim.x * blockDim.x) {
        int e = (int)(i >> 2), h = (int)(i & 3), s, d;
        edge_sd(ei, e, s, d);
        float sc = AS[s * HH + h] + AD[d * HH + h];
        sc = sc > 0.f ? sc : 0.2f * sc;
        float ex = __expf(sc - fdec(MX[d * HH + h]));
        EX[i] = ex;
        atomicAdd(&DEN[d * HH + h], ex);
    }
}
__global__ void k_edge_agg(const int* __restrict__ ei,
                           const float* __restrict__ EX, const float* __restrict__ DEN,
                           const float* __restrict__ Hf, float* __restrict__ AGG) {
    for (long i = blockIdx.x * (long)blockDim.x + threadIdx.x; i < (long)ETOT * HH;
         i += (long)gridDim.x * blockDim.x) {
        int e = (int)(i >> 2), h = (int)(i & 3), s, d;
        edge_sd(ei, e, s, d);
        float alpha = EX[i] / (DEN[d * HH + h] + 1e-16f);
        const float* hs = Hf + s * DD + h * CC;
        float* ag = AGG + d * DD + h * CC;
        #pragma unroll
        for (int c = 0; c < CC; ++c) atomicAdd(&ag[c], alpha * hs[c]);
    }
}

// ================= GAT post: x = LN(x + relu(agg + b)) (in place) =================
__global__ __launch_bounds__(128) void k_gat_post(float* __restrict__ X,
                                                  const float* __restrict__ AGG,
                                                  const float* __restrict__ gb,
                                                  const float* __restrict__ g,
                                                  const float* __restrict__ bb) {
    __shared__ float red[128];
    int n = blockIdx.x, c = threadIdx.x;
    float v = X[n * DD + c] + fmaxf(AGG[n * DD + c] + gb[c], 0.f);
    X[n * DD + c] = ln128(v, g, bb, red);
}

// ================= flash cross-attention, 1 wave per (16-query tile, head) =================
__global__ __launch_bounds__(32) void k_flash(const float* __restrict__ Q,
                                              const float* __restrict__ K,
                                              const float* __restrict__ V,
                                              float* __restrict__ XA) {
    const int qt = blockIdx.x;          // 0..383
    const int h  = blockIdx.y;          // 0..3
    const int lane = threadIdx.x;
    const int m = lane & 15, half = lane >> 4;
    const float SCALE = 0.17677669529663689f; // 1/sqrt(32)
    __shared__ float sP[16 * 32];

    // persistent A fragment: 16 query rows x 32 channels, pre-scaled
    u16v16 qa;
    const float* qrow = Q + (qt * 16 + m) * DD + h * CC;
    #pragma unroll
    for (int j = 0; j < 8; ++j) {
        int k0 = ((j < 4) ? 0 : 16) + half * 8 + 2 * (j & 3);
        qa[2 * j]     = f2bf(qrow[k0]     * SCALE);
        qa[2 * j + 1] = f2bf(qrow[k0 + 1] * SCALE);
    }

    v8f o0 = {0.f,0.f,0.f,0.f,0.f,0.f,0.f,0.f};
    v8f o1 = o0;
    float mrow[8], lrow[8];
    #pragma unroll
    for (int r = 0; r < 8; ++r) { mrow[r] = -__builtin_inff(); lrow[r] = 0.f; }

    for (int kb = 0; kb < NN; kb += 32) {
        if (kb + 64 < NN) {
            __builtin_prefetch(&K[(kb + 64) * DD + h * CC], 0, 1);  // global_prefetch_b8
            __builtin_prefetch(&V[(kb + 64) * DD + h * CC], 0, 1);
        }
        // B fragments for scores: lane = key (m), slots = channels (table order)
        u16v16 kb0, kb1;
        #pragma unroll
        for (int n = 0; n < 16; ++n) {
            int kk = ((n < 8) ? 0 : 16) + half * 8 + (n & 7);
            kb0[n] = f2bf(K[(kb + m)      * DD + h * CC + kk]);
            kb1[n] = f2bf(K[(kb + 16 + m) * DD + h * CC + kk]);
        }
        v8f zero = {0.f,0.f,0.f,0.f,0.f,0.f,0.f,0.f};
        v8f s0 = wmma_bf16(qa, kb0, zero);
        v8f s1 = wmma_bf16(qa, kb1, zero);

        // online softmax (row stats via half-wave butterfly)
        #pragma unroll
        for (int r = 0; r < 8; ++r) {
            float mx = fmaxf(s0[r], s1[r]);
            for (int off = 1; off < 16; off <<= 1) mx = fmaxf(mx, __shfl_xor(mx, off, 32));
            float mnew = fmaxf(mrow[r], mx);
            float p0 = __expf(s0[r] - mnew);
            float p1 = __expf(s1[r] - mnew);
            float rs = p0 + p1;
            for (int off = 1; off < 16; off <<= 1) rs += __shfl_xor(rs, off, 32);
            float corr = __expf(mrow[r] - mnew);
            lrow[r] = lrow[r] * corr + rs;
            mrow[r] = mnew;
            int rowl = r + half * 8;
            sP[rowl * 32 + m]      = p0;
            sP[rowl * 32 + 16 + m] = p1;
            o0[r] *= corr; o1[r] *= corr;
        }
        __syncthreads();

        // P (16x32) as A fragment
        u16v16 pa;
        #pragma unroll
        for (int j = 0; j < 8; ++j) {
            int k0 = ((j < 4) ? 0 : 16) + half * 8 + 2 * (j & 3);
            pa[2 * j]     = f2bf(sP[m * 32 + k0]);
            pa[2 * j + 1] = f2bf(sP[m * 32 + k0 + 1]);
        }
        // V (32 keys x 32 channels) as two B fragments: lane = channel, slots = keys (table order)
        u16v16 vb0, vb1;
        const float* vrow = V + kb * DD + h * CC + m;
        #pragma unroll
        for (int n = 0; n < 16; ++n) { int kk = ((n < 8) ? 0 : 16) + half * 8 + (n & 7); vb0[n] = f2bf(vrow[kk * DD]); vb1[n] = f2bf(vrow[kk * DD + 16]); }
        o0 = wmma_bf16(pa, vb0, o0);
        o1 = wmma_bf16(pa, vb1, o1);
        __syncthreads();
    }

    #pragma unroll
    for (int r = 0; r < 8; ++r) {
        int row = qt * 16 + r + half * 8;
        float inv = 1.f / (lrow[r] + 1e-16f);
        XA[row * DD + h * CC + m]      = o0[r] * inv;
        XA[row * DD + h * CC + 16 + m] = o1[r] * inv;
    }
}

// ================= CA post: xout = LN(x + xo) =================
__global__ __launch_bounds__(128) void k_ca_post(const float* __restrict__ X,
                                                 const float* __restrict__ XO,
                                                 const float* __restrict__ g,
                                                 const float* __restrict__ bb,
                                                 float* __restrict__ Xout) {
    __shared__ float red[128];
    int n = blockIdx.x, c = threadIdx.x;
    float v = X[n * DD + c] + XO[n * DD + c];
    Xout[n * DD + c] = ln128(v, g, bb, red);
}

// ================= policy final 64 -> 2 =================
__global__ void k_policy(const float* __restrict__ P2, const float* __restrict__ W3,
                         const float* __restrict__ b3, float* __restrict__ out) {
    int n = blockIdx.x * blockDim.x + threadIdx.x;
    if (n >= NQQ) return;
    float a0 = b3[0], a1 = b3[1];
    #pragma unroll
    for (int k = 0; k < 64; ++k) {
        float p = P2[n * 64 + k];
        a0 += p * W3[k * 2];
        a1 += p * W3[k * 2 + 1];
    }
    out[n * 2] = a0; out[n * 2 + 1] = a1;
}

// ================= mean pool over N rows =================
__global__ __launch_bounds__(128) void k_pool(const float* __restrict__ X, float* __restrict__ P) {
    int c = threadIdx.x;
    float s = 0.f;
    for (int n = 0; n < NN; ++n) s += X[n * DD + c];
    P[c] = s * (1.f / (float)NN);
}

// ================= value head (single block) =================
__global__ __launch_bounds__(128) void k_value(const float* __restrict__ P,
                                               const float* __restrict__ vW1, const float* __restrict__ vb1,
                                               const float* __restrict__ vW2, const float* __restrict__ vb2,
                                               const float* __restrict__ vW3, const float* __restrict__ vb3,
                                               float* __restrict__ out) {
    __shared__ float sp[128], s1[128], s2[64];
    int c = threadIdx.x;
    sp[c] = P[c]; __syncthreads();
    float a = vb1[c];
    for (int k = 0; k < 128; ++k) a += sp[k] * vW1[k * 128 + c];
    s1[c] = fmaxf(a, 0.f); __syncthreads();
    if (c < 64) {
        float a2 = vb2[c];
        for (int k = 0; k < 128; ++k) a2 += s1[k] * vW2[k * 64 + c];
        s2[c] = fmaxf(a2, 0.f);
    }
    __syncthreads();
    if (c == 0) {
        float v = vb3[0];
        for (int k = 0; k < 64; ++k) v += s2[k] * vW3[k];
        out[0] = v;
    }
}

// =============================== driver ===============================
extern "C" void kernel_launch(void* const* d_in, const int* in_sizes, int n_in,
                              void* d_out, int out_size, void* d_ws, size_t ws_size,
                              hipStream_t stream) {
    const float* nf    = (const float*)d_in[0];
    const int*   ei    = (const int*)  d_in[1];
    const float* pe    = (const float*)d_in[2];
    const float* emb_W = (const float*)d_in[3];
    const float* emb_b = (const float*)d_in[4];
    const float* elng  = (const float*)d_in[5];
    const float* elnb  = (const float*)d_in[6];
    const float* gatW  = (const float*)d_in[7];
    const float* asrc  = (const float*)d_in[8];
    const float* adst  = (const float*)d_in[9];
    const float* gatb  = (const float*)d_in[10];
    const float* lng   = (const float*)d_in[11];
    const float* lnb   = (const float*)d_in[12];
    const float* Wq    = (const float*)d_in[13];
    const float* Wk    = (const float*)d_in[14];
    const float* Wv    = (const float*)d_in[15];
    const float* Wo    = (const float*)d_in[16];
    const float* bq    = (const float*)d_in[17];
    const float* bk    = (const float*)d_in[18];
    const float* bv    = (const float*)d_in[19];
    const float* bo    = (const float*)d_in[20];
    const float* cag   = (const float*)d_in[21];
    const float* cab   = (const float*)d_in[22];
    const float* pW1   = (const float*)d_in[23];
    const float* pb1   = (const float*)d_in[24];
    const float* pW2   = (const float*)d_in[25];
    const float* pb2   = (const float*)d_in[26];
    const float* pW3   = (const float*)d_in[27];
    const float* pb3   = (const float*)d_in[28];
    const float* vW1   = (const float*)d_in[29];
    const float* vb1   = (const float*)d_in[30];
    const float* vW2   = (const float*)d_in[31];
    const float* vb2   = (const float*)d_in[32];
    const float* vW3   = (const float*)d_in[33];
    const float* vb3   = (const float*)d_in[34];

    // workspace layout (floats)
    float* ws  = (float*)d_ws;
    const int ND = NN * DD;                 // 786432
    float*    X    = ws;                    // node features
    float*    HB   = ws + ND;               // h / xa / p1
    float*    AGG  = ws + 2 * ND;           // agg / xo / p2
    float*    AS   = ws + 3 * ND;           // N*H
    float*    AD   = AS + NN * HH;
    unsigned* MX   = (unsigned*)(AD + NN * HH);
    float*    DEN  = AD + 2 * NN * HH;
    float*    POOL = DEN + NN * HH;         // 128 (+pad)
    float*    EX   = POOL + 256;            // ETOT*H floats; reused as Q,K after GAT
    float*    Qb   = EX;
    float*    Kb   = EX + ND;
    float*    Vb   = EX + (long)ETOT * HH;  // 786432 more

    float* dout   = (float*)d_out;
    float* logits = dout;            // 3072*2
    float* valptr = dout + NQQ * 2;  // 1
    float* Xout   = dout + NQQ * 2 + 1;  // final x (N*D), read in place by heads

    // 1) embedding
    k_embed<<<NN, 128, 0, stream>>>(nf, emb_W, emb_b, elng, elnb, X);

    // 2) GAT layers
    for (int l = 0; l < LL; ++l) {
        k_gemm<128><<<NN / 64, 128, 0, stream>>>(X, gatW + l * DD * DD, nullptr, HB, 0);
        k_scores<<<(NN * HH + 127) / 128, 128, 0, stream>>>(HB, asrc + l * HH * CC,
                                                            adst + l * HH * CC, AS, AD);
        k_fill_u<<<64, 256, 0, stream>>>(MX, FENC_NEG_INF, NN * HH);
        k_fill_f<<<64, 256, 0, stream>>>(DEN, 0.f, NN * HH);
        k_fill_f<<<1024, 256, 0, stream>>>(AGG, 0.f, ND);
        k_edge_max<<<2048, 256, 0, stream>>>(ei, AS, AD, MX);
        k_edge_expsum<<<2048, 256, 0, stream>>>(ei, AS, AD, MX, EX, DEN);
        k_edge_agg<<<2048, 256, 0, stream>>>(ei, EX, DEN, HB, AGG);
        k_gat_post<<<NN, 128, 0, stream>>>(X, AGG, gatb + l * DD, lng + l * DD, lnb + l * DD);
    }

    // 3) cross-attention q/k/v projections (EX region is free now)
    k_gemm<128><<<NN / 64, 128, 0, stream>>>(X,  Wq, bq, Qb, 0);
    k_gemm<128><<<NN / 64, 128, 0, stream>>>(pe, Wk, bk, Kb, 0);
    k_gemm<128><<<NN / 64, 128, 0, stream>>>(pe, Wv, bv, Vb, 0);

    // 4) flash attention -> HB (xa)
    {
        dim3 grid(NN / 16, HH, 1);
        k_flash<<<grid, 32, 0, stream>>>(Qb, Kb, Vb, HB);
    }

    // 5) output projection + residual LN -> Xout (lives in d_out)
    k_gemm<128><<<NN / 64, 128, 0, stream>>>(HB, Wo, bo, AGG, 0);
    k_ca_post<<<NN, 128, 0, stream>>>(X, AGG, cag, cab, Xout);

    // 6) policy head on first NQ rows of Xout
    k_gemm<128><<<NQQ / 64, 128, 0, stream>>>(Xout, pW1, pb1, HB, 1);
    k_gemm<64><<<NQQ / 64, 128, 0, stream>>>(HB, pW2, pb2, AGG, 1);
    k_policy<<<(NQQ + 255) / 256, 256, 0, stream>>>(AGG, pW3, pb3, logits);

    // 7) value head
    k_pool<<<1, 128, 0, stream>>>(Xout, POOL);
    k_value<<<1, 128, 0, stream>>>(POOL, vW1, vb1, vW2, vb2, vW3, vb3, valptr);

    (void)in_sizes; (void)n_in; (void)out_size; (void)ws_size;
}
